// Wordwise_47588237640315
// MI455X (gfx1250) — compile-verified, matrix operand order checked by reading, not run on hardware
//
#include <hip/hip_runtime.h>
#include <hip/hip_bf16.h>

// ---------------------------------------------------------------------------
// Wordwise pipeline for MI455X (gfx1250), bf16 WMMA implicit-GEMM convs,
// with TDM (tensor_load_to_lds) double-buffered weight staging.
// B=32, T=2048, MELS=80, H=128, K=5, MAX_WORDS=64, MAX_DUR=128.
// ---------------------------------------------------------------------------

typedef __attribute__((ext_vector_type(16))) __bf16 v16bf;
typedef __attribute__((ext_vector_type(8)))  float  v8f;
typedef unsigned int v4u __attribute__((ext_vector_type(4)));
typedef int          v8i __attribute__((ext_vector_type(8)));
typedef int          v4i __attribute__((ext_vector_type(4)));

union Frag {
    v16bf v;
    int4  q[2];
};

#define BATCH 32
#define TLEN  2048
#define MELS  80
#define CPAD1 96      // MELS padded to 3 k-steps of 32
#define HCH   128
#define NW    64      // MAX_WORDS
#define ND    128     // MAX_DUR

#if defined(__has_builtin)
#if __has_builtin(__builtin_amdgcn_tensor_load_to_lds)
#define WW_HAVE_TDM 1
#endif
#endif

#ifdef WW_HAVE_TDM
// 1-D TDM copy: nelem bf16 elements from global -> LDS (contiguous).
// D# per CDNA5 ISA ch.8: group0 = {count=1, lds_addr, global_addr, type=2},
// group1 = {data_size=1(2B), tensor_dim0=nelem, tile_dim0=nelem, stride0=nelem},
// groups 2/3 zero (tensor <= 2D). Tracked on TENSORcnt.
// This toolchain's builtin is the 6-arg form:
//   (uint32x4 g0, int32x8 g1, int32x4 g2, int32x4 g3, int32x8 pad, i32 cpol)
__device__ __forceinline__ void ww_tdm_issue(unsigned lds_addr, const void* gptr, unsigned nelem)
{
    unsigned long long ga = (unsigned long long)(size_t)gptr;
    v4u g0;
    g0[0] = 1u;                                   // count=1, user descriptor
    g0[1] = lds_addr;                             // LDS byte address (tile start)
    g0[2] = (unsigned)(ga & 0xffffffffu);         // global_addr[31:0]
    g0[3] = (unsigned)(ga >> 32) | (2u << 30);    // global_addr[56:32] | type=2
    v8i g1;
    g1[0] = (int)(1u << 16);                      // data_size=1 -> 2 bytes/elem
    g1[1] = (int)(nelem << 16);                   // tensor_dim0[15:0] (bits 63:48)
    g1[2] = 0;                                    // tensor_dim0 hi / tensor_dim1 lo
    g1[3] = (int)(nelem << 16);                   // tile_dim0 (bits 127:112)
    g1[4] = 0;                                    // tile_dim1=0 (unused), tile_dim2=0
    g1[5] = (int)nelem;                           // tensor_dim0_stride[31:0]
    g1[6] = 0;
    g1[7] = 0;
    v4i z4 = {0, 0, 0, 0};
    v8i z8 = {0, 0, 0, 0, 0, 0, 0, 0};
    __builtin_amdgcn_tensor_load_to_lds(g0, g1, z4, z4, z8, 0);
}
#endif

// ---------------------------------------------------------------------------
// Weight pre-pack: f32 [Cout=128][Cin][taps] -> bf16 per-lane fragment order
// packed[((tap*ksteps + ks)*8 + ntile)*32*16 + lane*16 + j]
// j<8  : k_local = (lane>=16 ? 8 : 0)  + j
// j>=8 : k_local = (lane>=16 ? 24 : 16) + (j-8)
// ---------------------------------------------------------------------------
__global__ void ww_pack_weights(const float* __restrict__ w, __bf16* __restrict__ packed,
                                int Cin, int taps, int ksteps, int total)
{
    int idx = blockIdx.x * blockDim.x + threadIdx.x;
    if (idx >= total) return;
    int j    = idx & 15;
    int lane = (idx >> 4) & 31;
    int nt   = (idx >> 9) & 7;
    int ks   = (idx >> 12) % ksteps;
    int tap  = idx / (4096 * ksteps);
    int hi   = (lane >> 4) & 1;
    int klocal = (j < 8) ? (hi * 8 + j) : (hi * 8 + 16 + (j - 8));
    int cin  = ks * 32 + klocal;
    int n    = nt * 16 + (lane & 15);
    float val = (cin < Cin) ? w[((size_t)n * Cin + cin) * taps + tap] : 0.0f;
    packed[idx] = (__bf16)val;
}

// ---------------------------------------------------------------------------
// features (B, MELS, T) f32 -> channels-last bf16 (B, T, CPAD1), zero-padded c
// ---------------------------------------------------------------------------
__global__ void ww_transpose_feat(const float* __restrict__ f, __bf16* __restrict__ ft)
{
    int idx = blockIdx.x * blockDim.x + threadIdx.x;
    int total = BATCH * TLEN * CPAD1;
    if (idx >= total) return;
    int c = idx % CPAD1;
    int t = (idx / CPAD1) % TLEN;
    int b = idx / (CPAD1 * TLEN);
    float val = (c < MELS) ? f[((size_t)b * MELS + c) * TLEN + t] : 0.0f;
    ft[idx] = (__bf16)val;
}

// ---------------------------------------------------------------------------
// conv1d (SAME, K=5) as implicit GEMM with bf16 WMMA.
// in  : (B, T, cinPad) bf16 channels-last
// out : (B, T, 128)    bf16 channels-last, bias + optional ReLU fused
// grid (T/128, B), block 256 (8 waves, 16 t-positions each)
// Weights: TDM double-buffered LDS staging (tap t+1 DMA overlaps tap t compute).
// ---------------------------------------------------------------------------
__global__ void ww_conv1d_wmma(const __bf16* __restrict__ in, int cinPad, int ksteps,
                               const __bf16* __restrict__ wp, const float* __restrict__ bias,
                               __bf16* __restrict__ out, int relu)
{
    __shared__ __bf16 wsm[2 * 16384];  // 2 x 32 KB weight double-buffer
    __shared__ __bf16 osm[8 * 2048];   // 8 waves x (16 x 128) bf16 output staging
    const int b    = blockIdx.y;
    const int tblk = blockIdx.x * 128;
    const int wave = threadIdx.x >> 5;
    const int lane = threadIdx.x & 31;
    const int hi   = (lane >> 4) & 1;
    const int lm   = lane & 15;
    const int t0   = tblk + wave * 16;
    const unsigned nelem = (unsigned)(ksteps * 4096);

    v8f acc[8] = {};

#ifdef WW_HAVE_TDM
    if (threadIdx.x < 32) {
        ww_tdm_issue((unsigned)(size_t)&wsm[0], wp, nelem);
        __builtin_amdgcn_s_wait_tensorcnt(0);
    }
    __syncthreads();
#endif

    for (int tap = 0; tap < 5; ++tap) {
        __bf16* cur = wsm + (tap & 1) * 16384;
#ifdef WW_HAVE_TDM
        if (threadIdx.x < 32 && tap < 4)   // async prefetch of next tap's weights
            ww_tdm_issue((unsigned)(size_t)(wsm + ((tap + 1) & 1) * 16384),
                         wp + (size_t)(tap + 1) * nelem, nelem);
#else
        cur = wsm;
        __syncthreads();
        {
            const int4* src = (const int4*)(wp + (size_t)tap * nelem);
            int4* dst = (int4*)cur;
            for (int i = 0; i < 2 * ksteps; ++i)
                dst[threadIdx.x + i * 256] = src[threadIdx.x + i * 256];
            if (tap < 4)
                __builtin_prefetch(wp + (size_t)(tap + 1) * nelem, 0, 0);
        }
        __syncthreads();
#endif

        const int tt = t0 + lm + tap - 2;
        const bool inb = (tt >= 0) && (tt < TLEN);

        for (int ks = 0; ks < ksteps; ++ks) {
            Frag a;
            if (inb) {
                const __bf16* ap = in + ((size_t)b * TLEN + tt) * cinPad + (ks * 32 + hi * 8);
                a.q[0] = *(const int4*)(ap);
                a.q[1] = *(const int4*)(ap + 16);
            } else {
                a.q[0] = make_int4(0, 0, 0, 0);
                a.q[1] = make_int4(0, 0, 0, 0);
            }

            const __bf16* base = cur + ks * 4096 + lane * 16;
#pragma unroll
            for (int nt = 0; nt < 8; nt += 2) {
                Frag b0, b1;
                const int4* bp0 = (const int4*)(base + nt * 512);
                const int4* bp1 = (const int4*)(base + (nt + 1) * 512);
                b0.q[0] = bp0[0]; b0.q[1] = bp0[1];
                b1.q[0] = bp1[0]; b1.q[1] = bp1[1];
                acc[nt] = __builtin_amdgcn_wmma_f32_16x16x32_bf16(
                    false, a.v, false, b0.v, (short)0, acc[nt], false, false);
                acc[nt + 1] = __builtin_amdgcn_wmma_f32_16x16x32_bf16(
                    false, a.v, false, b1.v, (short)0, acc[nt + 1], false, false);
            }
        }

#ifdef WW_HAVE_TDM
        if (threadIdx.x < 32 && tap < 4)
            __builtin_amdgcn_s_wait_tensorcnt(0);
        __syncthreads();   // next buffer filled, this buffer's readers done
#endif
    }

    // epilogue: bias + relu -> LDS staging -> coalesced b128 global stores
    __bf16* o = osm + wave * 2048;
#pragma unroll
    for (int nt = 0; nt < 8; ++nt) {
        const int n = nt * 16 + lm;
        const float bs = bias[n];
#pragma unroll
        for (int r = 0; r < 8; ++r) {
            float v = acc[nt][r] + bs;
            if (relu) v = fmaxf(v, 0.0f);
            o[(r + hi * 8) * HCH + n] = (__bf16)v;
        }
    }
    {
        const int4* s4 = (const int4*)o;
        int4* d4 = (int4*)(out + ((size_t)b * TLEN + t0) * HCH);
#pragma unroll
        for (int i = 0; i < 8; ++i)
            d4[lane + i * 32] = s4[lane + i * 32];
    }
}

// ---------------------------------------------------------------------------
// _slice_words: emb (B,T,128) bf16 -> padded (B,NW,ND,128) bf16 with validity mask
// grid (ND, NW, B), block 128 (= channels)
// ---------------------------------------------------------------------------
__global__ void ww_gather_words(const __bf16* __restrict__ emb,
                                const int* __restrict__ wb, const int* __restrict__ wl,
                                __bf16* __restrict__ padded)
{
    const int c = threadIdx.x;
    const int d = blockIdx.x;
    const int w = blockIdx.y;
    const int b = blockIdx.z;
    const int start = wb[b * 2 * NW + w];
    const int end   = wb[b * 2 * NW + NW + w];
    const bool valid = (d < (end - start)) && (w < wl[b]);
    int t = start + d;
    t = t < 0 ? 0 : (t > TLEN - 1 ? TLEN - 1 : t);
    __bf16 v = valid ? emb[((size_t)b * TLEN + t) * HCH + c] : (__bf16)0.0f;
    padded[(((size_t)b * NW + w) * ND + d) * HCH + c] = v;
}

// ---------------------------------------------------------------------------
// conv2d 5x5 SAME, 128->128 ch, as implicit GEMM with bf16 WMMA.
// in/out: (B, NW, ND, 128) bf16 channels-last. bias + ReLU fused.
// grid (NW, B), block 256 (8 waves x 16 d-positions = full d row of 128)
// K = 128 cin x 25 taps = 3200 -> per tap: 4 k-steps of 32 = 32 KB weights,
// TDM double-buffered so the DMA of tap t+1 overlaps tap t's 32 WMMAs.
// ---------------------------------------------------------------------------
__global__ void ww_conv2d_wmma(const __bf16* __restrict__ in,
                               const __bf16* __restrict__ wp, const float* __restrict__ bias,
                               __bf16* __restrict__ out)
{
    __shared__ __bf16 wsm[2 * 16384];  // 2 x 32 KB weight double-buffer
    __shared__ __bf16 osm[8 * 2048];   // 32 KB output staging
    const int b    = blockIdx.y;
    const int w    = blockIdx.x;
    const int wave = threadIdx.x >> 5;
    const int lane = threadIdx.x & 31;
    const int hi   = (lane >> 4) & 1;
    const int lm   = lane & 15;
    const int d0   = wave * 16;

    v8f acc[8] = {};

#ifdef WW_HAVE_TDM
    if (threadIdx.x < 32) {
        ww_tdm_issue((unsigned)(size_t)&wsm[0], wp, 16384u);
        __builtin_amdgcn_s_wait_tensorcnt(0);
    }
    __syncthreads();
#endif

    for (int tap = 0; tap < 25; ++tap) {
        __bf16* cur = wsm + (tap & 1) * 16384;
#ifdef WW_HAVE_TDM
        if (threadIdx.x < 32 && tap < 24)
            ww_tdm_issue((unsigned)(size_t)(wsm + ((tap + 1) & 1) * 16384),
                         wp + (size_t)(tap + 1) * 16384, 16384u);
#else
        cur = wsm;
        __syncthreads();
        {
            const int4* src = (const int4*)(wp + (size_t)tap * 16384);
            int4* dst = (int4*)cur;
#pragma unroll
            for (int i = 0; i < 8; ++i)
                dst[threadIdx.x + i * 256] = src[threadIdx.x + i * 256];
            if (tap < 24)
                __builtin_prefetch(wp + (size_t)(tap + 1) * 16384, 0, 0);
        }
        __syncthreads();
#endif

        const int dw = tap / 5 - 2;
        const int dd = tap % 5 - 2;
        const int wr = w + dw;
        const int din = d0 + lm + dd;
        const bool inb = (wr >= 0) && (wr < NW) && (din >= 0) && (din < ND);

        for (int ks = 0; ks < 4; ++ks) {
            Frag a;
            if (inb) {
                const __bf16* ap = in + (((size_t)b * NW + wr) * ND + din) * HCH
                                      + (ks * 32 + hi * 8);
                a.q[0] = *(const int4*)(ap);
                a.q[1] = *(const int4*)(ap + 16);
            } else {
                a.q[0] = make_int4(0, 0, 0, 0);
                a.q[1] = make_int4(0, 0, 0, 0);
            }

            const __bf16* base = cur + ks * 4096 + lane * 16;
#pragma unroll
            for (int nt = 0; nt < 8; nt += 2) {
                Frag b0, b1;
                const int4* bp0 = (const int4*)(base + nt * 512);
                const int4* bp1 = (const int4*)(base + (nt + 1) * 512);
                b0.q[0] = bp0[0]; b0.q[1] = bp0[1];
                b1.q[0] = bp1[0]; b1.q[1] = bp1[1];
                acc[nt] = __builtin_amdgcn_wmma_f32_16x16x32_bf16(
                    false, a.v, false, b0.v, (short)0, acc[nt], false, false);
                acc[nt + 1] = __builtin_amdgcn_wmma_f32_16x16x32_bf16(
                    false, a.v, false, b1.v, (short)0, acc[nt + 1], false, false);
            }
        }

#ifdef WW_HAVE_TDM
        if (threadIdx.x < 32 && tap < 24)
            __builtin_amdgcn_s_wait_tensorcnt(0);
        __syncthreads();
#endif
    }

    // epilogue: bias + relu -> LDS -> coalesced b128 stores (wave tile is 4 KB contiguous)
    __bf16* o = osm + wave * 2048;
#pragma unroll
    for (int nt = 0; nt < 8; ++nt) {
        const int n = nt * 16 + lm;
        const float bs = bias[n];
#pragma unroll
        for (int r = 0; r < 8; ++r) {
            float v = fmaxf(acc[nt][r] + bs, 0.0f);
            o[(r + hi * 8) * HCH + n] = (__bf16)v;
        }
    }
    {
        const int4* s4 = (const int4*)o;
        int4* d4 = (int4*)(out + (((size_t)b * NW + w) * ND + d0) * HCH);
#pragma unroll
        for (int i = 0; i < 8; ++i)
            d4[lane + i * 32] = s4[lane + i * 32];
    }
}

// ---------------------------------------------------------------------------
// einsum('bcwd,od->bcwo') with lw (1,128) + lb + ReLU  -> z (B, NW, 128) f32
// grid (NW, B), block 128 (= channels)
// ---------------------------------------------------------------------------
__global__ void ww_reduce_dur(const __bf16* __restrict__ y,
                              const float* __restrict__ lw, const float* __restrict__ lb,
                              float* __restrict__ z)
{
    const int c = threadIdx.x;
    const int w = blockIdx.x;
    const int b = blockIdx.y;
    const __bf16* p = y + (((size_t)b * NW + w) * ND) * HCH + c;
    float s = 0.0f;
#pragma unroll 4
    for (int d = 0; d < ND; ++d)
        s += (float)p[(size_t)d * HCH] * lw[d];
    s += lb[0];
    z[((size_t)b * NW + w) * HCH + c] = fmaxf(s, 0.0f);
}

// ---------------------------------------------------------------------------
// Final conv2d: (B,128,NW,1) -> (B,1,NW,1). Width-1 dim => only kw=2 column.
// out[b,w] = cb3 + sum_{i,c} z[b][w+i-2][c] * cw3[0][c][i][2]
// grid (NW, B), block 128 with LDS tree reduction.
// ---------------------------------------------------------------------------
__global__ void ww_final_conv(const float* __restrict__ z,
                              const float* __restrict__ cw3, const float* __restrict__ cb3,
                              float* __restrict__ out)
{
    __shared__ float red[128];
    const int c = threadIdx.x;
    const int w = blockIdx.x;
    const int b = blockIdx.y;
    float s = 0.0f;
#pragma unroll
    for (int i = 0; i < 5; ++i) {
        const int wr = w + i - 2;
        if (wr >= 0 && wr < NW)
            s += z[((size_t)b * NW + wr) * HCH + c] * cw3[(c * 5 + i) * 5 + 2];
    }
    red[c] = s;
    __syncthreads();
    for (int off = 64; off > 0; off >>= 1) {
        if (c < off) red[c] += red[c + off];
        __syncthreads();
    }
    if (c == 0) out[b * NW + w] = red[0] + cb3[0];
}

// ---------------------------------------------------------------------------
extern "C" void kernel_launch(void* const* d_in, const int* in_sizes, int n_in,
                              void* d_out, int out_size, void* d_ws, size_t ws_size,
                              hipStream_t stream)
{
    const float* features = (const float*)d_in[0];
    const int*   wbounds  = (const int*)d_in[1];
    const int*   wlens    = (const int*)d_in[2];
    const float* w1  = (const float*)d_in[3];
    const float* b1  = (const float*)d_in[4];
    const float* w2  = (const float*)d_in[5];
    const float* b2  = (const float*)d_in[6];
    const float* w3  = (const float*)d_in[7];
    const float* b3  = (const float*)d_in[8];
    const float* cw1 = (const float*)d_in[9];
    const float* cb1 = (const float*)d_in[10];
    const float* cw2 = (const float*)d_in[11];
    const float* cb2 = (const float*)d_in[12];
    const float* lw  = (const float*)d_in[13];
    const float* lb  = (const float*)d_in[14];
    const float* cw3 = (const float*)d_in[15];
    const float* cb3 = (const float*)d_in[16];
    float* out = (float*)d_out;

    // workspace layout (bytes)
    char* ws = (char*)d_ws;
    __bf16* pw1   = (__bf16*)(ws + 0);            //   122,880 B
    __bf16* pw2   = (__bf16*)(ws + 122880);       //   163,840 B
    __bf16* pw3   = (__bf16*)(ws + 286720);       //   163,840 B
    __bf16* pcw1  = (__bf16*)(ws + 450560);       //   819,200 B
    __bf16* pcw2  = (__bf16*)(ws + 1269760);      //   819,200 B
    __bf16* featT = (__bf16*)(ws + 2088960);      // 12,582,912 B
    __bf16* x1    = (__bf16*)(ws + 14671872);     // 16,777,216 B  (also reused as emb)
    __bf16* x2    = (__bf16*)(ws + 31449088);     // 16,777,216 B
    __bf16* bufP  = (__bf16*)(ws + 48226304);     // 67,108,864 B  (padded, then y2)
    __bf16* bufQ  = (__bf16*)(ws + 115335168);    // 67,108,864 B  (y1)
    float*  zbuf  = (float*)(ws + 182444032);     //  1,048,576 B

    // 1) pack weights to bf16 fragment order
    ww_pack_weights<<<dim3((61440 + 255) / 256), dim3(256), 0, stream>>>(w1, pw1, MELS, 5, 3, 61440);
    ww_pack_weights<<<dim3((81920 + 255) / 256), dim3(256), 0, stream>>>(w2, pw2, HCH, 5, 4, 81920);
    ww_pack_weights<<<dim3((81920 + 255) / 256), dim3(256), 0, stream>>>(w3, pw3, HCH, 5, 4, 81920);
    ww_pack_weights<<<dim3((409600 + 255) / 256), dim3(256), 0, stream>>>(cw1, pcw1, HCH, 25, 4, 409600);
    ww_pack_weights<<<dim3((409600 + 255) / 256), dim3(256), 0, stream>>>(cw2, pcw2, HCH, 25, 4, 409600);

    // 2) features -> channels-last bf16, c padded to 96
    {
        int total = BATCH * TLEN * CPAD1;
        ww_transpose_feat<<<dim3((total + 255) / 256), dim3(256), 0, stream>>>(features, featT);
    }

    // 3) conv1d stack (WMMA)
    ww_conv1d_wmma<<<dim3(TLEN / 128, BATCH), dim3(256), 0, stream>>>(featT, CPAD1, 3, pw1, b1, x1, 1);
    ww_conv1d_wmma<<<dim3(TLEN / 128, BATCH), dim3(256), 0, stream>>>(x1, HCH, 4, pw2, b2, x2, 1);
    ww_conv1d_wmma<<<dim3(TLEN / 128, BATCH), dim3(256), 0, stream>>>(x2, HCH, 4, pw3, b3, x1, 0); // emb -> x1

    // 4) word slicing with validity mask
    ww_gather_words<<<dim3(ND, NW, BATCH), dim3(128), 0, stream>>>(x1, wbounds, wlens, bufP);

    // 5) two 5x5 conv2d layers (WMMA) -- ~430 GFLOP, the hot loop
    ww_conv2d_wmma<<<dim3(NW, BATCH), dim3(256), 0, stream>>>(bufP, pcw1, cb1, bufQ);
    ww_conv2d_wmma<<<dim3(NW, BATCH), dim3(256), 0, stream>>>(bufQ, pcw2, cb2, bufP);

    // 6) duration contraction (lw) + bias + ReLU
    ww_reduce_dur<<<dim3(NW, BATCH), dim3(128), 0, stream>>>(bufP, lw, lb, zbuf);

    // 7) final width-1 conv2d -> (B,1,NW,1)
    ww_final_conv<<<dim3(NW, BATCH), dim3(128), 0, stream>>>(zbuf, cw3, cb3, out);
}
